// Attention_2826088481284
// MI455X (gfx1250) — matrix operand order repaired and checked by reading, not compiled, not run on hardware
//
#include <hip/hip_runtime.h>

// ---------------------------------------------------------------------------
// Problem constants (reference: B=2, S=4096, D=1024, H=16, W=128, HD=64)
// ---------------------------------------------------------------------------
#define BSZ   2
#define SEQ   4096
#define DMODEL 1024
#define NHEAD 16
#define HDIM  64
#define WIN   128
#define NBLK  (SEQ / WIN)       // 32
#define MROWS (BSZ * SEQ)       // 8192

typedef __attribute__((ext_vector_type(16))) __bf16 v16bf;
typedef __attribute__((ext_vector_type(8)))  float  v8f;
typedef __attribute__((ext_vector_type(4)))  int    v4i;

// ---- CDNA5 async global->LDS path (guarded; falls back to sync copy) ------
#if defined(__has_builtin)
#if __has_builtin(__builtin_amdgcn_global_load_async_to_lds_b128) && \
    __has_builtin(__builtin_amdgcn_s_wait_asynccnt)
#define USE_ASYNC_LDS 1
#endif
#endif
#ifndef USE_ASYNC_LDS
#define USE_ASYNC_LDS 0
#endif

static __device__ __forceinline__ void copy16_g2l(const __bf16* gsrc, __bf16* ldst) {
#if USE_ASYNC_LDS
  // Builtin signature (probed via diagnostics): (v4i AS1*, v4i AS3*, imm, imm)
  __builtin_amdgcn_global_load_async_to_lds_b128(
      (__attribute__((address_space(1))) v4i*)gsrc,
      (__attribute__((address_space(3))) v4i*)ldst, 0, 0);
#else
  *reinterpret_cast<float4*>(ldst) = *reinterpret_cast<const float4*>(gsrc);
#endif
}

static __device__ __forceinline__ void async_drain() {
#if USE_ASYNC_LDS
  __builtin_amdgcn_s_wait_asynccnt(0);
#endif
}

static __device__ __forceinline__ v8f wmma_bf16(v16bf a, v16bf b, v8f c) {
  return __builtin_amdgcn_wmma_f32_16x16x32_bf16(
      /*neg_a=*/false, a, /*neg_b=*/false, b,
      /*c_mod=*/(short)0, c, /*reuse_a=*/false, /*reuse_b=*/false);
}

// Load a v16bf fragment as two 16-byte chunks (LDS -> ds_load_b128,
// global -> global_load_b128). p0/p1 must be 16B aligned.
static __device__ __forceinline__ v16bf ld_pair(const __bf16* p0, const __bf16* p1) {
  union { v16bf v; float4 f[2]; } u;
  u.f[0] = *reinterpret_cast<const float4*>(p0);
  u.f[1] = *reinterpret_cast<const float4*>(p1);
  return u.v;
}

// Convert 8 fp32 -> 8 bf16 and store as one 16B chunk.
static __device__ __forceinline__ void cvt8_store(__bf16* dst, const float* s) {
  union { __bf16 e[8]; float4 f; } u;
#pragma unroll
  for (int i = 0; i < 8; ++i) u.e[i] = (__bf16)s[i];
  *reinterpret_cast<float4*>(dst) = u.f;
}

// ---------------------------------------------------------------------------
// Tiled GEMM:  C[m,n] = sum_k A[m,k] * W[n,k]   (i.e. A @ W^T)
//   AF32: A is fp32 (converted to bf16 while staging), else A is bf16
//         (staged with async global->LDS copies).
//   OF32: output fp32 (final projection to d_out), else bf16 (+ optional RoPE).
// Block tile 128x128, BK=32, 256 threads = 8 wave32 in a 4(M) x 2(N) grid;
// each wave owns a 32x64 subtile = 2x4 WMMA tiles. The 64-wide wave N-subtile
// is head-aligned so RoPE pairs (d, d+32) sit in acc tiles (j, j+2).
// grid.x = numWeights*8 N-tiles, grid.y = 64 M-tiles.
// ---------------------------------------------------------------------------
template <bool AF32, bool OF32>
__global__ __launch_bounds__(256) void gemm_wmma(
    const void* __restrict__ Av,
    const float* __restrict__ w0, const float* __restrict__ w1,
    const float* __restrict__ w2,
    void* __restrict__ o0, void* __restrict__ o1, void* __restrict__ o2,
    const float* __restrict__ rope) {
  __shared__ __bf16 sA[128 * 40];   // 128 rows x 32 (+8 pad) bf16
  __shared__ __bf16 sB[128 * 40];

  const int nt    = blockIdx.x & 7;       // N tile index within this weight
  const int which = blockIdx.x >> 3;      // 0..2 selects weight/output
  const int m0    = blockIdx.y * 128;
  const int n0    = nt * 128;

  const float* Wt = (which == 0) ? w0 : (which == 1) ? w1 : w2;
  void*        Ov = (which == 0) ? o0 : (which == 1) ? o1 : o2;

  const int t    = threadIdx.x;
  const int lane = t & 31;
  const int wave = t >> 5;
  const int wr   = wave & 3;              // M sub-row (0..3) * 32
  const int wc   = wave >> 2;             // N sub-col (0..1) * 64
  const int lh   = (lane >> 4) & 1;       // lane half (A/B K-interleave)
  const int ln   = lane & 15;

  const int srow = t >> 1;                // staging: row 0..127
  const int scol = (t & 1) * 16;          // staging: 16-element half-row

  v8f acc[2][4];
#pragma unroll
  for (int mi = 0; mi < 2; ++mi)
#pragma unroll
    for (int ni = 0; ni < 4; ++ni) acc[mi][ni] = (v8f)0.0f;

  for (int k0 = 0; k0 < DMODEL; k0 += 32) {
    __syncthreads();
    // ---- stage A tile (convert to bf16 if fp32, else async copy) ----
    if (AF32) {
      const float* a = (const float*)Av + (size_t)(m0 + srow) * DMODEL + k0 + scol;
      union { float4 f[4]; float s[16]; } ub;
      ub.f[0] = *reinterpret_cast<const float4*>(a + 0);
      ub.f[1] = *reinterpret_cast<const float4*>(a + 4);
      ub.f[2] = *reinterpret_cast<const float4*>(a + 8);
      ub.f[3] = *reinterpret_cast<const float4*>(a + 12);
      cvt8_store(sA + srow * 40 + scol + 0, ub.s + 0);
      cvt8_store(sA + srow * 40 + scol + 8, ub.s + 8);
      if (k0 + 32 < DMODEL) __builtin_prefetch(a + 32, 0, 1);
    } else {
      const __bf16* a = (const __bf16*)Av + (size_t)(m0 + srow) * DMODEL + k0 + scol;
      copy16_g2l(a + 0, sA + srow * 40 + scol + 0);
      copy16_g2l(a + 8, sA + srow * 40 + scol + 8);
      if (k0 + 32 < DMODEL) __builtin_prefetch(a + 32, 0, 1);
    }
    // ---- stage B tile = rows of W (fp32 -> bf16) ----
    {
      const float* b = Wt + (size_t)(n0 + srow) * DMODEL + k0 + scol;
      union { float4 f[4]; float s[16]; } ub;
      ub.f[0] = *reinterpret_cast<const float4*>(b + 0);
      ub.f[1] = *reinterpret_cast<const float4*>(b + 4);
      ub.f[2] = *reinterpret_cast<const float4*>(b + 8);
      ub.f[3] = *reinterpret_cast<const float4*>(b + 12);
      cvt8_store(sB + srow * 40 + scol + 0, ub.s + 0);
      cvt8_store(sB + srow * 40 + scol + 8, ub.s + 8);
    }
    if (!AF32) async_drain();   // ASYNCcnt must be 0 before barrier signal
    __syncthreads();

    // ---- fragments & WMMA ----
    v16bf af[2], bfr[4];
#pragma unroll
    for (int mi = 0; mi < 2; ++mi) {
      const __bf16* p = sA + (wr * 32 + mi * 16 + ln) * 40 + lh * 8;
      af[mi] = ld_pair(p, p + 16);                 // A: chunks at K {0,16}+8*lh
    }
#pragma unroll
    for (int ni = 0; ni < 4; ++ni) {
      const __bf16* p = sB + (wc * 64 + ni * 16 + ln) * 40 + lh * 8;
      bfr[ni] = ld_pair(p, p + 16);                // B: chunks at K {0,16}+8*lh
    }
#pragma unroll
    for (int mi = 0; mi < 2; ++mi)
#pragma unroll
      for (int ni = 0; ni < 4; ++ni)
        acc[mi][ni] = wmma_bf16(af[mi], bfr[ni], acc[mi][ni]);
  }

  // ---- epilogue ----
  const int nbase = n0 + wc * 64;
  if (OF32) {
    float* O = (float*)Ov;
#pragma unroll
    for (int mi = 0; mi < 2; ++mi)
#pragma unroll
      for (int ni = 0; ni < 4; ++ni)
#pragma unroll
        for (int r = 0; r < 8; ++r) {
          const int m = m0 + wr * 32 + mi * 16 + r + lh * 8;
          O[(size_t)m * DMODEL + nbase + ni * 16 + ln] = acc[mi][ni][r];
        }
  } else {
    __bf16* O = (__bf16*)Ov;
    const bool doRope = (which < 2);   // q,k get RoPE; v does not
    if (doRope) {
#pragma unroll
      for (int mi = 0; mi < 2; ++mi)
#pragma unroll
        for (int pj = 0; pj < 2; ++pj) {           // pairs (0,2) and (1,3)
          const int fi = pj * 16 + ln;             // freq index 0..31
#pragma unroll
          for (int r = 0; r < 8; ++r) {
            const int m = m0 + wr * 32 + mi * 16 + r + lh * 8;
            const int s = m & (SEQ - 1);
            const float f  = rope[(size_t)s * (HDIM / 2) + fi];
            const float cf = __cosf(f);
            const float sf = __sinf(f);
            const float t1 = acc[mi][pj][r];
            const float t2 = acc[mi][pj + 2][r];
            const int n1 = nbase + pj * 16 + ln;
            O[(size_t)m * DMODEL + n1]      = (__bf16)(t1 * cf - t2 * sf);
            O[(size_t)m * DMODEL + n1 + 32] = (__bf16)(t2 * cf + t1 * sf);
          }
        }
    } else {
#pragma unroll
      for (int mi = 0; mi < 2; ++mi)
#pragma unroll
        for (int ni = 0; ni < 4; ++ni)
#pragma unroll
          for (int r = 0; r < 8; ++r) {
            const int m = m0 + wr * 32 + mi * 16 + r + lh * 8;
            O[(size_t)m * DMODEL + nbase + ni * 16 + ln] = (__bf16)acc[mi][ni][r];
          }
    }
  }
}

// ---------------------------------------------------------------------------
// Flash-style sliding-window attention.
// One workgroup per (batch, query-block of 128, head). 8 waves x 16 rows.
// Loop over the 3 key blocks {nq-1, nq, nq+1}: stage K (async global->LDS)
// and V^T in LDS, S = Q K^T via WMMA, mask + online softmax (shfl_xor row
// reductions in 16-lane halves), P -> per-wave LDS slab -> A-fragments,
// O += P V via WMMA.  grid = (NHEAD, NBLK, BSZ), block = 256.
// ---------------------------------------------------------------------------
__global__ __launch_bounds__(256) void attn_wmma(
    const void* __restrict__ qv, const void* __restrict__ kv,
    const void* __restrict__ vv, void* __restrict__ ov) {
  __shared__ __bf16 sK[128 * 72];        // K block: 128 keys x 64 (+8 pad)
  __shared__ __bf16 sVT[64 * 136];       // V^T:     64 dims x 128 keys (+8 pad)
  __shared__ __bf16 sP[8 * 16 * 136];    // per-wave probs: 16 rows x 128 (+8)

  const __bf16* qg = (const __bf16*)qv;
  const __bf16* kg = (const __bf16*)kv;
  const __bf16* vg = (const __bf16*)vv;
  __bf16*       og = (__bf16*)ov;

  const int h  = blockIdx.x;
  const int nq = blockIdx.y;
  const int b  = blockIdx.z;

  const int t    = threadIdx.x;
  const int lane = t & 31;
  const int wave = t >> 5;
  const int lh   = (lane >> 4) & 1;
  const int ln   = lane & 15;

  // ---- load Q fragments straight from global (A layout) ----
  const int qrow = nq * WIN + wave * 16 + ln;
  const __bf16* qp =
      qg + ((size_t)(b * SEQ + qrow) * DMODEL + h * HDIM);
  v16bf aq[2];
#pragma unroll
  for (int tc = 0; tc < 2; ++tc) {
    const __bf16* p = qp + tc * 32 + lh * 8;
    aq[tc] = ld_pair(p, p + 16);
  }

  v8f oacc[4];
#pragma unroll
  for (int no = 0; no < 4; ++no) oacc[no] = (v8f)0.0f;
  float mrun[8], lrun[8];
#pragma unroll
  for (int r = 0; r < 8; ++r) { mrun[r] = -3.0e38f; lrun[r] = 0.0f; }

  for (int j = 0; j < 3; ++j) {
    const int ks = (nq + j - 1) * WIN;   // may be negative / past end
    __syncthreads();                      // previous-iter LDS reads done

    // ---- stage K block via async global->LDS (zero-fill OOB rows) ----
    {
      const int row = t >> 1, cb = (t & 1) * 32;
      const int pos = ks + row;
      __bf16* dst = sK + row * 72 + cb;
      if (pos >= 0 && pos < SEQ) {
        const __bf16* src = kg + ((size_t)(b * SEQ + pos) * DMODEL + h * HDIM + cb);
#pragma unroll
        for (int i = 0; i < 4; ++i) copy16_g2l(src + i * 8, dst + i * 8);
      } else {
        const float4 z = make_float4(0.f, 0.f, 0.f, 0.f);
#pragma unroll
        for (int i = 0; i < 4; ++i) *reinterpret_cast<float4*>(dst + i * 8) = z;
      }
    }
    // ---- stage V^T (transpose while writing, zero-fill OOB) ----
    {
      const int kk = t >> 1, db = (t & 1) * 32;
      const int pos = ks + kk;
      union { float4 f[4]; __bf16 e[32]; } uv;
      if (pos >= 0 && pos < SEQ) {
        const __bf16* src = vg + ((size_t)(b * SEQ + pos) * DMODEL + h * HDIM + db);
#pragma unroll
        for (int i = 0; i < 4; ++i)
          uv.f[i] = *reinterpret_cast<const float4*>(src + i * 8);
      } else {
        const float4 z = make_float4(0.f, 0.f, 0.f, 0.f);
#pragma unroll
        for (int i = 0; i < 4; ++i) uv.f[i] = z;
      }
#pragma unroll
      for (int d = 0; d < 32; ++d) sVT[(db + d) * 136 + kk] = uv.e[d];
    }
    async_drain();                        // ASYNCcnt == 0 before barrier
    __syncthreads();

    // ---- scores S = Q K^T (8 N-tiles of 16 keys, K=64 in two chunks) ----
    v8f sc[8];
#pragma unroll
    for (int ni = 0; ni < 8; ++ni) {
      sc[ni] = (v8f)0.0f;
#pragma unroll
      for (int tc = 0; tc < 2; ++tc) {
        const __bf16* p = sK + (ni * 16 + ln) * 72 + tc * 32 + lh * 8;
        sc[ni] = wmma_bf16(aq[tc], ld_pair(p, p + 16), sc[ni]);
      }
    }

    // ---- scale + window mask ----
    const int qpos0 = nq * WIN + wave * 16 + lh * 8;
#pragma unroll
    for (int ni = 0; ni < 8; ++ni) {
      const int kpos = ks + ni * 16 + ln;
      const bool kvalid = (kpos >= 0) && (kpos < SEQ);
#pragma unroll
      for (int r = 0; r < 8; ++r) {
        const int rel = kpos - (qpos0 + r);
        const bool ok = kvalid && (rel >= -(WIN / 2)) && (rel <= (WIN / 2 - 1));
        sc[ni][r] = ok ? sc[ni][r] * 0.125f : -1.0e30f;
      }
    }

    // ---- online softmax: row max / rescale / exp / row sum ----
    float alpha[8];
#pragma unroll
    for (int r = 0; r < 8; ++r) {
      float mx = -3.0e38f;
#pragma unroll
      for (int ni = 0; ni < 8; ++ni) mx = fmaxf(mx, sc[ni][r]);
#pragma unroll
      for (int msk = 1; msk < 16; msk <<= 1)
        mx = fmaxf(mx, __shfl_xor(mx, msk, 32));   // 16-lane halves
      const float nm = fmaxf(mrun[r], mx);
      alpha[r] = __expf(mrun[r] - nm);
      mrun[r] = nm;
    }
#pragma unroll
    for (int ni = 0; ni < 8; ++ni)
#pragma unroll
      for (int r = 0; r < 8; ++r) sc[ni][r] = __expf(sc[ni][r] - mrun[r]);
#pragma unroll
    for (int r = 0; r < 8; ++r) {
      float sm = 0.0f;
#pragma unroll
      for (int ni = 0; ni < 8; ++ni) sm += sc[ni][r];
#pragma unroll
      for (int msk = 1; msk < 16; msk <<= 1) sm += __shfl_xor(sm, msk, 32);
      lrun[r] = lrun[r] * alpha[r] + sm;
#pragma unroll
      for (int no = 0; no < 4; ++no) oacc[no][r] *= alpha[r];
    }

    // ---- P to per-wave LDS slab (re-layout into A fragments) ----
    __bf16* pb = sP + wave * (16 * 136);
#pragma unroll
    for (int ni = 0; ni < 8; ++ni)
#pragma unroll
      for (int r = 0; r < 8; ++r)
        pb[(r + lh * 8) * 136 + ni * 16 + ln] = (__bf16)sc[ni][r];

    // ---- O += P V  (M=16, K=128 in 4 chunks, N=64 in 4 tiles) ----
#pragma unroll
    for (int no = 0; no < 4; ++no)
#pragma unroll
      for (int kc = 0; kc < 4; ++kc) {
        const __bf16* pa = pb + ln * 136 + kc * 32 + lh * 8;
        const __bf16* pv = sVT + (no * 16 + ln) * 136 + kc * 32 + lh * 8;
        oacc[no] = wmma_bf16(ld_pair(pa, pa + 16), ld_pair(pv, pv + 16), oacc[no]);
      }
  }

  // ---- epilogue: normalize and store bf16 ----
#pragma unroll
  for (int no = 0; no < 4; ++no)
#pragma unroll
    for (int r = 0; r < 8; ++r) {
      const int row = nq * WIN + wave * 16 + r + lh * 8;
      og[(size_t)(b * SEQ + row) * DMODEL + h * HDIM + no * 16 + ln] =
          (__bf16)(oacc[no][r] / lrun[r]);
    }
}

// ---------------------------------------------------------------------------
// Host launcher.
// Inputs: x(f32 B*S*D), rope_freqs(f32 S*32), wq, wk, wv, wo (f32 D*D).
// Output: f32 B*S*D. Workspace: q,k,v,attn_out as bf16 (4 x 16 MiB).
// ---------------------------------------------------------------------------
extern "C" void kernel_launch(void* const* d_in, const int* in_sizes, int n_in,
                              void* d_out, int out_size, void* d_ws, size_t ws_size,
                              hipStream_t stream) {
  (void)in_sizes; (void)n_in; (void)out_size; (void)ws_size;
  const float* x    = (const float*)d_in[0];
  const float* rope = (const float*)d_in[1];
  const float* wq   = (const float*)d_in[2];
  const float* wk   = (const float*)d_in[3];
  const float* wv   = (const float*)d_in[4];
  const float* wo   = (const float*)d_in[5];
  float* out = (float*)d_out;

  const size_t elems = (size_t)MROWS * DMODEL;      // 8M bf16 = 16 MiB each
  char* ws = (char*)d_ws;
  void* qb = ws;
  void* kb = ws + 2 * elems;
  void* vb = ws + 4 * elems;
  void* ab = ws + 6 * elems;

  // 1) q,k,v = x @ w{q,k,v}^T  (+RoPE on q,k), bf16 out
  gemm_wmma<true, false><<<dim3(24, MROWS / 128), 256, 0, stream>>>(
      x, wq, wk, wv, qb, kb, vb, rope);

  // 2) sliding-window attention
  attn_wmma<<<dim3(NHEAD, NBLK, BSZ), 256, 0, stream>>>(qb, kb, vb, ab);

  // 3) out = attn @ wo^T, fp32 out
  gemm_wmma<false, true><<<dim3(8, MROWS / 128), 256, 0, stream>>>(
      ab, wo, wo, wo, out, out, out, rope);
}
